// SFE_2001454760195
// MI455X (gfx1250) — compile-verified, matrix operand order checked by reading, not run on hardware
//
#include <hip/hip_runtime.h>

typedef __attribute__((ext_vector_type(16))) __bf16 v16bf;
typedef __attribute__((ext_vector_type(8)))  __bf16 v8bf;
typedef __attribute__((ext_vector_type(8)))  float  v8f;
typedef __attribute__((ext_vector_type(4)))  float  v4f;

#define CH 64
#define NEG_SLOPE 0.01f
#define GN_EPS 1e-5f

// ---------------------------------------------------------------------------
// zero kernels
// ---------------------------------------------------------------------------
__global__ __launch_bounds__(256) void sfe_zero4(float4* __restrict__ p, int n4) {
  int i = blockIdx.x * blockDim.x + threadIdx.x;
  if (i < n4) p[i] = make_float4(0.f, 0.f, 0.f, 0.f);
}

__global__ void sfe_zero_stats(float* __restrict__ p) {
  if (threadIdx.x < 16) p[threadIdx.x] = 0.f;
}

// ---------------------------------------------------------------------------
// gather -> bf16 WMMA GEMM -> scatter-add (one wave = 16 rows x 64 cols)
// feats:[*,64] fp32, W:[K,64,64] fp32, maps:[K,M], out:[Nout,64] fp32 accum
// ---------------------------------------------------------------------------
__global__ __launch_bounds__(256) void sfe_conv_gemm_scatter(
    const float* __restrict__ feats, const float* __restrict__ W,
    const int* __restrict__ in_map, const int* __restrict__ out_map,
    float* __restrict__ out, int M)
{
  __shared__ __align__(16) __bf16 wlds[CH * CH];  // [n][k] bf16, 8 KB

  const int k   = blockIdx.y;
  const int tid = threadIdx.x;

  // ---- stage W[k] transposed ([kk][n] fp32 -> [n][kk] bf16) into LDS ----
  {
    const float* wk = W + (size_t)k * CH * CH;
    const int n   = tid & 63;          // output channel (coalesced across lanes)
    const int kk0 = (tid >> 6) * 16;   // 16 consecutive input channels
    v8bf lo, hi;
#pragma unroll
    for (int i = 0; i < 8; ++i) lo[i] = (__bf16)wk[(kk0 + i) * CH + n];
#pragma unroll
    for (int i = 0; i < 8; ++i) hi[i] = (__bf16)wk[(kk0 + 8 + i) * CH + n];
    v8bf* dst = (v8bf*)&wlds[n * CH + kk0];
    dst[0] = lo;
    dst[1] = hi;
  }
  __syncthreads();

  const int wave   = tid >> 5;
  const int lane   = tid & 31;
  const int Mtiles = (M + 15) >> 4;
  const int tile   = blockIdx.x * 8 + wave;
  if (tile >= Mtiles) return;          // whole-wave exit; EXEC stays full below

  const int m0   = tile * 16;
  const int half = lane >> 4;          // 0: lanes 0-15, 1: lanes 16-31
  const int lrow = lane & 15;
  const int kM   = k * M;

  // gather row for this lane's matrix row (clamped; OOB rows masked at store)
  int mm = m0 + lrow;
  if (mm > M - 1) mm = M - 1;
  const int grow = in_map[kM + mm];
  const float* arow = feats + (size_t)grow * CH + half * 8;

  v8f acc[4];
#pragma unroll
  for (int t = 0; t < 4; ++t) acc[t] = (v8f){};

#pragma unroll
  for (int s = 0; s < 2; ++s) {        // K = s*32 .. s*32+31
    // A operand: ISA 16-bit A layout -> runs K = s*32+half*8+{0..7} and +16
    v4f a0 = *(const v4f*)(arow + s * 32);
    v4f a1 = *(const v4f*)(arow + s * 32 + 4);
    v4f a2 = *(const v4f*)(arow + s * 32 + 16);
    v4f a3 = *(const v4f*)(arow + s * 32 + 20);
    v16bf A;
#pragma unroll
    for (int i = 0; i < 4; ++i) {
      A[i]      = (__bf16)a0[i];
      A[4 + i]  = (__bf16)a1[i];
      A[8 + i]  = (__bf16)a2[i];
      A[12 + i] = (__bf16)a3[i];
    }
#pragma unroll
    for (int t = 0; t < 4; ++t) {      // 16 output columns per WMMA
      const __bf16* bp = &wlds[(t * 16 + lrow) * CH + s * 32 + half * 8];
      v8bf b0 = *(const v8bf*)bp;
      v8bf b1 = *(const v8bf*)(bp + 16);
      v16bf B;
#pragma unroll
      for (int i = 0; i < 8; ++i) { B[i] = b0[i]; B[8 + i] = b1[i]; }
      acc[t] = __builtin_amdgcn_wmma_f32_16x16x32_bf16(
          false, A, false, B, (short)0, acc[t], false, false);
    }
  }

  // ---- scatter-add: lane holds D rows M=half*8+j, col lrow+16t ----
#pragma unroll
  for (int j = 0; j < 8; ++j) {
    const int m = m0 + half * 8 + j;
    if (m < M) {
      const int orow = out_map[kM + m];
      float* op = out + (size_t)orow * CH + lrow;
      unsafeAtomicAdd(op +  0, acc[0][j]);
      unsafeAtomicAdd(op + 16, acc[1][j]);
      unsafeAtomicAdd(op + 32, acc[2][j]);
      unsafeAtomicAdd(op + 48, acc[3][j]);
    }
  }
}

// ---------------------------------------------------------------------------
// GroupNorm stats: stats[0..7]=sum per group, stats[8..15]=sumsq per group
// total must be a multiple of 64; thread's channel (idx&63) is loop-invariant
// ---------------------------------------------------------------------------
__global__ __launch_bounds__(256) void sfe_gn_stats(
    const float* __restrict__ x, float* __restrict__ stats, int total)
{
  __shared__ float ssum[8], ssq[8];
  if (threadIdx.x < 8) { ssum[threadIdx.x] = 0.f; ssq[threadIdx.x] = 0.f; }
  __syncthreads();

  const int idx    = blockIdx.x * blockDim.x + threadIdx.x;
  const int stride = gridDim.x * blockDim.x;   // multiple of 64
  const int g      = (idx & 63) >> 3;
  float s = 0.f, q = 0.f;
  for (int i = idx; i < total; i += stride) {
    float v = x[i];
    s += v;
    q += v * v;
  }
  atomicAdd(&ssum[g], s);
  atomicAdd(&ssq[g], q);
  __syncthreads();
  if (threadIdx.x < 8) {
    unsafeAtomicAdd(&stats[threadIdx.x], ssum[threadIdx.x]);
    unsafeAtomicAdd(&stats[8 + threadIdx.x], ssq[threadIdx.x]);
  }
}

// ---------------------------------------------------------------------------
// apply GN (+ optional residual) (+ optional leaky)
// mode: 0 = GN only, 1 = GN+leaky, 2 = GN+residual+leaky
// ---------------------------------------------------------------------------
__global__ __launch_bounds__(256) void sfe_gn_apply(
    const float* __restrict__ y, const float* __restrict__ stats,
    const float* __restrict__ gamma, const float* __restrict__ beta,
    const float* __restrict__ resid, float* __restrict__ outp,
    int total, float inv_cnt, int mode)
{
  const int idx    = blockIdx.x * blockDim.x + threadIdx.x;
  const int stride = gridDim.x * blockDim.x;
  for (int i = idx; i < total; i += stride) {
    const int c = i & 63;
    const int g = c >> 3;
    const float mean = stats[g] * inv_cnt;
    const float var  = stats[8 + g] * inv_cnt - mean * mean;
    const float rs   = rsqrtf(var + GN_EPS);
    float v = (y[i] - mean) * rs * gamma[c] + beta[c];
    if (mode == 2) v += resid[i];
    if (mode != 0) v = (v >= 0.f) ? v : NEG_SLOPE * v;
    outp[i] = v;
  }
}

// ---------------------------------------------------------------------------
// host-side phase: zero accum -> conv -> stats -> apply
// ---------------------------------------------------------------------------
static void sfe_phase(const float* fin, const float* W, const int* imap,
                      const int* omap, int K, int M,
                      const float* gamma, const float* beta,
                      const float* resid, float* outp,
                      float* y, float* stats, int Nout, int mode,
                      hipStream_t stream)
{
  const int total = Nout * CH;
  sfe_zero4<<<(total / 4 + 255) / 256, 256, 0, stream>>>((float4*)y, total / 4);
  sfe_zero_stats<<<1, 32, 0, stream>>>(stats);

  const int Mtiles = (M + 15) / 16;
  dim3 grid((Mtiles + 7) / 8, K);
  sfe_conv_gemm_scatter<<<grid, 256, 0, stream>>>(fin, W, imap, omap, y, M);

  sfe_gn_stats<<<512, 256, 0, stream>>>(y, stats, total);

  const float inv_cnt = 1.0f / ((float)Nout * 8.0f);
  sfe_gn_apply<<<1024, 256, 0, stream>>>(y, stats, gamma, beta, resid, outp,
                                         total, inv_cnt, mode);
}

extern "C" void kernel_launch(void* const* d_in, const int* in_sizes, int n_in,
                              void* d_out, int out_size, void* d_ws, size_t ws_size,
                              hipStream_t stream) {
  const float* x   = (const float*)d_in[0];
  const float* W1  = (const float*)d_in[1];
  const float* g1  = (const float*)d_in[2];
  const float* b1  = (const float*)d_in[3];
  const float* Wa1 = (const float*)d_in[4];
  const float* ga1 = (const float*)d_in[5];
  const float* ba1 = (const float*)d_in[6];
  const float* Wb1 = (const float*)d_in[7];
  const float* gb1 = (const float*)d_in[8];
  const float* bb1 = (const float*)d_in[9];
  const float* Wa2 = (const float*)d_in[10];
  const float* ga2 = (const float*)d_in[11];
  const float* ba2 = (const float*)d_in[12];
  const float* Wb2 = (const float*)d_in[13];
  const float* gb2 = (const float*)d_in[14];
  const float* bb2 = (const float*)d_in[15];
  const int* m1_in  = (const int*)d_in[16];
  const int* m1_out = (const int*)d_in[17];
  const int* m2_in  = (const int*)d_in[18];
  const int* m2_out = (const int*)d_in[19];

  const int M1   = in_sizes[16] / 8;    // 75000
  const int M2   = in_sizes[18] / 27;   // 60000
  const int Nout = out_size / CH;       // 150000

  float* h     = (float*)d_out;                      // persistent feature buf
  float* y     = (float*)d_ws;                       // conv accumulator
  float* t     = y + (size_t)Nout * CH;              // mid-residual buf
  float* stats = t + (size_t)Nout * CH;              // 16 floats

  // BasicConvolutionBlock: conv(ks=2) -> GN -> leaky          => h
  sfe_phase(x, W1, m1_in, m1_out, 8, M1, g1, b1, nullptr, h, y, stats, Nout, 1, stream);
  // ResidualBlock 1
  sfe_phase(h, Wa1, m2_in, m2_out, 27, M2, ga1, ba1, nullptr, t, y, stats, Nout, 0, stream);
  sfe_phase(t, Wb1, m2_in, m2_out, 27, M2, gb1, bb1, h, h, y, stats, Nout, 2, stream);
  // ResidualBlock 2
  sfe_phase(h, Wa2, m2_in, m2_out, 27, M2, ga2, ba2, nullptr, t, y, stats, Nout, 0, stream);
  sfe_phase(t, Wb2, m2_in, m2_out, 27, M2, gb2, bb2, h, h, y, stats, Nout, 2, stream);
}
